// Attention_21895743275585
// MI455X (gfx1250) — matrix operand order repaired and checked by reading, not compiled, not run on hardware
//
#include <hip/hip_runtime.h>
#include <hip/hip_bf16.h>

// ---------------------------------------------------------------------------
// Attention (l2-normalized q/k, scale=10) for MI455X / gfx1250, wave32 WMMA.
// All matrix math on v_wmma_f32_16x16x32_f16 (f16 operands, f32 accum).
// Flash-attention streaming: the 4x8x2304x2304 score matrix is never
// materialized (would be ~679MB of HBM traffic).
// ---------------------------------------------------------------------------

#define B_     4
#define CH     256
#define HW     2304      // 48*48
#define HEADS  8
#define DH     64
#define INNER  512       // HEADS*DH
#define OQKV   1536      // 3*INNER
#define SCALE_ 10.0f

typedef __attribute__((ext_vector_type(16))) _Float16 v16h;
typedef __attribute__((ext_vector_type(8)))  _Float16 v8h;
typedef __attribute__((ext_vector_type(8)))  float    v8f;

#define WMMA_F16(a,b,c) \
  __builtin_amdgcn_wmma_f32_16x16x32_f16(false,(a),false,(b),(short)0,(c),false,false)

// A operand (16x32, MxK) from row-major [M x ld] f16, tile at (row0, k0).
// ISA layout: lanes 0-15 hold row M=lane with K {0..7,16..23}; lanes 16-31
// hold the same rows with K {8..15,24..31}. Two 16B loads per lane.
static __device__ __forceinline__ v16h load_frag_a(const _Float16* base, int ld,
                                                   int row0, int k0) {
  int lane = threadIdx.x & 31;
  const _Float16* p = base + (size_t)(row0 + (lane & 15)) * ld
                           + (k0 + ((lane >> 4) << 3));
  v8h lo = *(const v8h*)(p);        // K = kb .. kb+7
  v8h hi = *(const v8h*)(p + 16);   // K = kb+16 .. kb+23
  v16h r;
#pragma unroll
  for (int i = 0; i < 8; ++i) { r[i] = lo[i]; r[i + 8] = hi[i]; }
  return r;
}

// B operand (32x16, KxN) taken from B^T stored row-major [N x ld] f16
// (row n of the buffer = column n of B, contiguous in K).
// Layout: lanes 0-15 hold column N=lane, K {0..7,16..23}; lanes 16-31 K {8..15,24..31}.
static __device__ __forceinline__ v16h load_frag_b(const _Float16* baseT, int ld,
                                                   int n0, int k0) {
  int lane = threadIdx.x & 31;
  const _Float16* p = baseT + (size_t)(n0 + (lane & 15)) * ld
                            + (k0 + ((lane >> 4) << 3));
  v8h lo = *(const v8h*)(p);
  v8h hi = *(const v8h*)(p + 16);
  v16h r;
#pragma unroll
  for (int i = 0; i < 8; ++i) { r[i] = lo[i]; r[i + 8] = hi[i]; }
  return r;
}

// ---------------- small prep kernels ----------------

__global__ void cvt_f32_to_f16(const float* __restrict__ src,
                               _Float16* __restrict__ dst, int n) {
  int i = blockIdx.x * 256 + threadIdx.x;
  if (i < n) dst[i] = (_Float16)src[i];
}

// x [b][c][n] f32 -> xT [b][n][c] f16 via 32x32 LDS tile (coalesced both ways).
// Grid: B_ * (CH/32) * (HW/32) blocks of 256 threads.
__global__ __launch_bounds__(256) void transpose_x_f16(
    const float* __restrict__ x, _Float16* __restrict__ xT) {
  __shared__ float tile[32][33];                 // padded: conflict-free
  int tx = threadIdx.x & 31;
  int ty = threadIdx.x >> 5;                     // 0..7
  int blk = blockIdx.x;
  int b  = blk / (8 * 72);
  int r  = blk % (8 * 72);
  int c0 = (r / 72) * 32;
  int n0 = (r % 72) * 32;
#pragma unroll
  for (int j = 0; j < 4; ++j) {
    int c = c0 + ty + j * 8;
    tile[ty + j * 8][tx] = x[((size_t)b * CH + c) * HW + n0 + tx];
  }
  __syncthreads();
#pragma unroll
  for (int j = 0; j < 4; ++j) {
    int n = n0 + ty + j * 8;
    xT[((size_t)b * HW + n) * CH + c0 + tx] = (_Float16)tile[tx][ty + j * 8];
  }
}

// ---------------- QKV projection GEMM ----------------
// Per batch: W[1536x256] * X[256x2304]. Each wave: 16x64 tile.
// q,k scattered to [b,h,n,d]; v scattered transposed to vT [b,h,d,n].
__global__ __launch_bounds__(256) void qkv_gemm(
    const _Float16* __restrict__ wq, const _Float16* __restrict__ xT,
    _Float16* __restrict__ qb, _Float16* __restrict__ kb,
    _Float16* __restrict__ vT) {
  int wid  = threadIdx.x >> 5;
  int lane = threadIdx.x & 31;
  int w    = blockIdx.x * 8 + wid;               // 4*96*36 waves exactly
  int b    = w / (96 * 36);
  int rem  = w % (96 * 36);
  int o0   = (rem / 36) * 16;
  int n0   = (rem % 36) * 64;
  const _Float16* xb = xT + (size_t)b * HW * CH;

  v8f acc[4] = {};
#pragma unroll
  for (int k0 = 0; k0 < CH; k0 += 32) {
    v16h a = load_frag_a(wq, CH, o0, k0);
#pragma unroll
    for (int j = 0; j < 4; ++j) {
      v16h bf = load_frag_b(xb, CH, n0 + j * 16, k0);
      acc[j] = WMMA_F16(a, bf, acc[j]);
    }
  }

  int hi8 = (lane >> 4) << 3;
  int col = lane & 15;
#pragma unroll
  for (int j = 0; j < 4; ++j)
#pragma unroll
    for (int i = 0; i < 8; ++i) {
      int o = o0 + i + hi8;
      int n = n0 + j * 16 + col;
      int part = o >> 9;            // 0=q 1=k 2=v (uniform per tile)
      int head = (o >> 6) & 7;
      int dd   = o & 63;
      _Float16 v = (_Float16)acc[j][i];
      size_t bh = (size_t)(b * HEADS + head);
      if (part == 0)      qb[(bh * HW + n) * DH + dd] = v;
      else if (part == 1) kb[(bh * HW + n) * DH + dd] = v;
      else                vT[(bh * DH + dd) * HW + n] = v;
    }
}

// ---------------- l2 normalize q,k rows (scale folded into q) -------------
// One thread per 64-element row; vectorized 16B loads/stores.
__global__ void l2norm_qk(_Float16* __restrict__ q, _Float16* __restrict__ k) {
  int row = blockIdx.x * 256 + threadIdx.x;      // exact: 4*8*2304 rows
  v8h* qr = (v8h*)(q + (size_t)row * DH);
  v8h* kr = (v8h*)(k + (size_t)row * DH);
  v8h qv[8], kv[8];
  float sq = 0.f, sk = 0.f;
#pragma unroll
  for (int j = 0; j < 8; ++j) {
    qv[j] = qr[j];
    kv[j] = kr[j];
#pragma unroll
    for (int i = 0; i < 8; ++i) {
      float a = (float)qv[j][i]; sq += a * a;
      float b = (float)kv[j][i]; sk += b * b;
    }
  }
  float iq = SCALE_ / fmaxf(sqrtf(sq), 1e-12f);
  float ik = 1.0f   / fmaxf(sqrtf(sk), 1e-12f);
#pragma unroll
  for (int j = 0; j < 8; ++j) {
    v8h qo, ko;
#pragma unroll
    for (int i = 0; i < 8; ++i) {
      qo[i] = (_Float16)((float)qv[j][i] * iq);
      ko[i] = (_Float16)((float)kv[j][i] * ik);
    }
    qr[j] = qo;
    kr[j] = ko;
  }
}

// ---------------- flash attention ----------------
// 576 blocks x 8 waves; wave owns 16 query rows, streams 32-key tiles.
// Row of the 16x16 C/D tile lives across one 16-lane half, so online-softmax
// row reductions are shfl_xor masks 1/2/4/8.
__global__ __launch_bounds__(256) void flash_attn(
    const _Float16* __restrict__ q, const _Float16* __restrict__ k,
    const _Float16* __restrict__ vT, _Float16* __restrict__ innerT) {
  __shared__ _Float16 plds[8][16 * 32];          // per-wave P staging (1KB)

  int wid  = threadIdx.x >> 5;
  int lane = threadIdx.x & 31;
  int bh   = blockIdx.x / 18;                    // (b,h) pair
  int rb   = blockIdx.x % 18;
  int b    = bh >> 3;
  int h    = bh & 7;
  int r0   = rb * 128 + wid * 16;                // query rows of this wave

  const _Float16* qbh = q  + (size_t)bh * HW * DH;
  const _Float16* kbh = k  + (size_t)bh * HW * DH;
  const _Float16* vbh = vT + (size_t)bh * DH * HW;

  v16h qa0 = load_frag_a(qbh, DH, r0, 0);
  v16h qa1 = load_frag_a(qbh, DH, r0, 32);

  v8f O[4] = {};
  float m[8], l[8];
#pragma unroll
  for (int i = 0; i < 8; ++i) { m[i] = -1e30f; l[i] = 0.f; }

  int hi8 = (lane >> 4) << 3;
  int col = lane & 15;
  _Float16* pl = plds[wid];

  for (int j0 = 0; j0 < HW; j0 += 32) {
    // prefetch next key/value tile into cache while this tile computes
    // (uniform branch: EXEC stays all-ones for the WMMAs below)
    if (j0 + 32 < HW) {
      // next K tile: 32 rows x 64 f16 = 4KB = 32 lanes x 128B
      __builtin_prefetch(kbh + (size_t)(j0 + 32) * DH + lane * 64, 0, 3);
      // next V tile: 64 d-rows x 32 keys (strided); one line per 2 d-rows
      __builtin_prefetch(vbh + (size_t)(lane * 2) * HW + j0 + 32, 0, 3);
    }

    // S = q . k^T   (16 rows x 32 keys), two 16-col accumulators
    v16h kb0 = load_frag_b(kbh, DH, j0, 0);
    v16h kb1 = load_frag_b(kbh, DH, j0, 32);
    v8f s0 = {};
    s0 = WMMA_F16(qa0, kb0, s0);
    s0 = WMMA_F16(qa1, kb1, s0);
    v16h kb2 = load_frag_b(kbh, DH, j0 + 16, 0);
    v16h kb3 = load_frag_b(kbh, DH, j0 + 16, 32);
    v8f s1 = {};
    s1 = WMMA_F16(qa0, kb2, s1);
    s1 = WMMA_F16(qa1, kb3, s1);

    // online softmax per row slot
    float p0[8], p1[8];
#pragma unroll
    for (int i = 0; i < 8; ++i) {
      float loc = fmaxf(s0[i], s1[i]);
      loc = fmaxf(loc, __shfl_xor(loc, 1));
      loc = fmaxf(loc, __shfl_xor(loc, 2));
      loc = fmaxf(loc, __shfl_xor(loc, 4));
      loc = fmaxf(loc, __shfl_xor(loc, 8));
      float mn = fmaxf(m[i], loc);
      float alpha = __expf(m[i] - mn);
      m[i] = mn;
      p0[i] = __expf(s0[i] - mn);
      p1[i] = __expf(s1[i] - mn);
      float rs = p0[i] + p1[i];
      rs += __shfl_xor(rs, 1);
      rs += __shfl_xor(rs, 2);
      rs += __shfl_xor(rs, 4);
      rs += __shfl_xor(rs, 8);
      l[i] = l[i] * alpha + rs;
#pragma unroll
      for (int t = 0; t < 4; ++t) O[t][i] *= alpha;
    }

    // re-swizzle P (C/D layout) -> A-fragment layout through per-wave LDS
#pragma unroll
    for (int i = 0; i < 8; ++i) {
      pl[(i + hi8) * 32 + col]      = (_Float16)p0[i];
      pl[(i + hi8) * 32 + 16 + col] = (_Float16)p1[i];
    }
    v16h pa = load_frag_a(pl, 32, 0, 0);

    // O += P . V   (V^T rows = d, contiguous in keys)
#pragma unroll
    for (int t = 0; t < 4; ++t) {
      v16h vb = load_frag_b(vbh, HW, t * 16, j0);
      O[t] = WMMA_F16(pa, vb, O[t]);
    }
  }

  // epilogue: O / l  ->  innerT[b][n][h*64 + d]  (f16, output-GEMM B operand)
#pragma unroll
  for (int i = 0; i < 8; ++i) {
    float inv = 1.0f / l[i];
    int n = r0 + i + hi8;
#pragma unroll
    for (int t = 0; t < 4; ++t) {
      innerT[((size_t)b * HW + n) * INNER + h * 64 + t * 16 + col] =
          (_Float16)(O[t][i] * inv);
    }
  }
}

// ---------------- output projection GEMM ----------------
// Per batch: Wout[256x512] * inner[512x2304] + bias -> out [b][o][n] f32.
__global__ __launch_bounds__(256) void out_gemm(
    const _Float16* __restrict__ wo, const _Float16* __restrict__ innerT,
    const float* __restrict__ bias, float* __restrict__ out) {
  int wid  = threadIdx.x >> 5;
  int lane = threadIdx.x & 31;
  int w    = blockIdx.x * 8 + wid;               // 4*16*36 waves exactly
  int b    = w / (16 * 36);
  int rem  = w % (16 * 36);
  int o0   = (rem / 36) * 16;
  int n0   = (rem % 36) * 64;
  const _Float16* ib = innerT + (size_t)b * HW * INNER;

  v8f acc[4] = {};
#pragma unroll
  for (int k0 = 0; k0 < INNER; k0 += 32) {
    v16h a = load_frag_a(wo, INNER, o0, k0);
#pragma unroll
    for (int j = 0; j < 4; ++j) {
      v16h bf = load_frag_b(ib, INNER, n0 + j * 16, k0);
      acc[j] = WMMA_F16(a, bf, acc[j]);
    }
  }

  int hi8 = (lane >> 4) << 3;
  int col = lane & 15;
#pragma unroll
  for (int j = 0; j < 4; ++j)
#pragma unroll
    for (int i = 0; i < 8; ++i) {
      int o = o0 + i + hi8;
      int n = n0 + j * 16 + col;
      out[((size_t)b * CH + o) * HW + n] = acc[j][i] + bias[o];
    }
}

// ---------------- launch ----------------

extern "C" void kernel_launch(void* const* d_in, const int* in_sizes, int n_in,
                              void* d_out, int out_size, void* d_ws, size_t ws_size,
                              hipStream_t stream) {
  const float* x     = (const float*)d_in[0];   // [4,256,48,48]
  const float* w_qkv = (const float*)d_in[1];   // [1536,256]
  const float* w_out = (const float*)d_in[2];   // [256,512]
  const float* b_out = (const float*)d_in[3];   // [256]
  float* out = (float*)d_out;

  // carve workspace (256B aligned)
  char* ws = (char*)d_ws;
  size_t off = 0;
  auto carve = [&](size_t bytes) -> char* {
    char* p = ws + off;
    off = (off + bytes + 255) & ~(size_t)255;
    return p;
  };
  _Float16* xT     = (_Float16*)carve((size_t)B_ * HW * CH * 2);      // 4.7MB
  _Float16* wqh    = (_Float16*)carve((size_t)OQKV * CH * 2);         // 0.8MB
  _Float16* woh    = (_Float16*)carve((size_t)CH * INNER * 2);        // 0.3MB
  _Float16* qbuf   = (_Float16*)carve((size_t)B_ * HEADS * HW * DH * 2); // 9.4MB
  _Float16* kbuf   = (_Float16*)carve((size_t)B_ * HEADS * HW * DH * 2); // 9.4MB
  _Float16* vTbuf  = (_Float16*)carve((size_t)B_ * HEADS * DH * HW * 2); // 9.4MB
  _Float16* innerT = (_Float16*)carve((size_t)B_ * HW * INNER * 2);      // 9.4MB

  // 1) weight converts + x transpose (tiled, coalesced)
  cvt_f32_to_f16<<<(OQKV * CH + 255) / 256, 256, 0, stream>>>(w_qkv, wqh, OQKV * CH);
  cvt_f32_to_f16<<<(CH * INNER + 255) / 256, 256, 0, stream>>>(w_out, woh, CH * INNER);
  transpose_x_f16<<<B_ * 8 * 72, 256, 0, stream>>>(x, xT);

  // 2) QKV projection: 4 * 96 * 36 waves, 8 waves/block
  qkv_gemm<<<(B_ * 96 * 36) / 8, 256, 0, stream>>>(wqh, xT, qbuf, kbuf, vTbuf);

  // 3) l2 normalize (scale folded into q)
  l2norm_qk<<<(B_ * HEADS * HW) / 256, 256, 0, stream>>>(qbuf, kbuf);

  // 4) flash attention: 32 (b,h) pairs * 18 row blocks of 128
  flash_attn<<<B_ * HEADS * 18, 256, 0, stream>>>(qbuf, kbuf, vTbuf, innerT);

  // 5) output projection + bias: 4 * 16 * 36 waves
  out_gemm<<<(B_ * 16 * 36) / 8, 256, 0, stream>>>(woh, innerT, b_out, out);
}
